// EncoderLayer_58007828299809
// MI455X (gfx1250) — matrix operand order repaired and checked by reading, not compiled, not run on hardware
//
#include <hip/hip_runtime.h>
#include <cstdint>
#include <cstddef>

// ---------------------------------------------------------------------------
// Problem constants (match reference): B=4, S=2048, D=1024, H=16, DK=64, DFF=4096
// ---------------------------------------------------------------------------
constexpr int cB = 4, cS = 2048, cD = 1024, cH = 16, cDK = 64, cDFF = 4096;
constexpr float cEPS = 1e-6f;
constexpr int cM = cB * cS;            // 8192 rows

typedef __attribute__((ext_vector_type(16))) __bf16        bf16x16;
typedef __attribute__((ext_vector_type(8)))  float         f32x8;
typedef __attribute__((ext_vector_type(8)))  unsigned int  u32x8;
typedef __attribute__((ext_vector_type(4)))  unsigned int  u32x4;

#define DEV static __device__ __forceinline__

// B-fragment (32x16 K-major, 16-bit): lane (half=lane>>4, n=lane&15) holds
// K = half*16 + {0..15} of column n  ->  16 contiguous bf16 (global or LDS).
DEV bf16x16 load_b_frag(const __bf16* p) {
    const uint4* q = (const uint4*)p;
    uint4 a = q[0], b = q[1];
    u32x8 u = {a.x, a.y, a.z, a.w, b.x, b.y, b.z, b.w};
    return __builtin_bit_cast(bf16x16, u);
}

// A-fragment (16x32, 16-bit): lane (half=lane>>4, m=lane&15) holds
// K = half*16 + {0..15}  ->  16 contiguous bf16.
DEV bf16x16 load_a_frag(const __bf16* p) {
    uint4 a = *(const uint4*)(p);
    uint4 b = *(const uint4*)(p + 8);
    u32x8 u = {a.x, a.y, a.z, a.w, b.x, b.y, b.z, b.w};
    return __builtin_bit_cast(bf16x16, u);
}

DEV f32x8 wmma_bf16(bf16x16 a, bf16x16 b, f32x8 c) {
    return __builtin_amdgcn_wmma_f32_16x16x32_bf16(
        false, a, false, b, (short)0, c, false, false);
}

// Generic pointer to LDS byte offset: flat LDS aperture keeps offset in [31:0].
DEV unsigned lds_addr_of(const void* p) { return (unsigned)(uintptr_t)p; }

// Async global->LDS copy, 16 bytes per lane. Tracked by ASYNCcnt.
DEV void async_load_b128(unsigned lds, const __bf16* g) {
    asm volatile("global_load_async_to_lds_b128 %0, %1, off"
                 :: "v"(lds), "v"((unsigned long long)(uintptr_t)g)
                 : "memory");
}
DEV void wait_async0() { asm volatile("s_wait_asynccnt 0x0" ::: "memory"); }

// Tensor Data Mover: 2D tile (rows x cols bf16, row stride in elements) from
// global to LDS. D# per CDNA5 ISA:
//  group0: [1:0]=count=1, [63:32]=lds_addr, [120:64]=global_addr, [127:126]=2
//  group1: data_size=1(2B), tensor_dim0/1, tile_dim0=cols, tile_dim1=rows,
//          tensor_dim0_stride=row_stride. Tracked by TENSORcnt.
DEV void tdm_load_2d(unsigned lds, const __bf16* g, unsigned rows, unsigned cols,
                     unsigned row_stride, unsigned tdim0, unsigned tdim1) {
    unsigned long long ga = (unsigned long long)(uintptr_t)g;
    u32x4 g0;
    g0[0] = 1u;                                               // count=1
    g0[1] = lds;                                              // lds_addr
    g0[2] = (unsigned)ga;                                     // global_addr lo
    g0[3] = ((unsigned)(ga >> 32) & 0x01FFFFFFu) | 0x80000000u; // addr hi | type=2
    u32x8 g1;
    g1[0] = 0x00010000u;                                      // data_size=1 (2B)
    g1[1] = (tdim0 & 0xFFFFu) << 16;                          // tensor_dim0 lo16
    g1[2] = (tdim0 >> 16) | ((tdim1 & 0xFFFFu) << 16);        // dim0 hi | dim1 lo
    g1[3] = (tdim1 >> 16) | (cols << 16);                     // dim1 hi | tile_dim0
    g1[4] = rows;                                             // tile_dim1, tile_dim2=0
    g1[5] = row_stride;                                       // dim0_stride lo32
    g1[6] = 0u;
    g1[7] = 0u;
    asm volatile("tensor_load_to_lds %0, %1" :: "s"(g0), "s"(g1) : "memory");
}

// ---------------------------------------------------------------------------
// fp32 -> bf16 cast (n always a multiple of 256)
// ---------------------------------------------------------------------------
__global__ __launch_bounds__(256) void cast_kernel(const float* __restrict__ in,
                                                   __bf16* __restrict__ out) {
    size_t i = (size_t)blockIdx.x * 256 + threadIdx.x;
    out[i] = (__bf16)in[i];
}

// ---------------------------------------------------------------------------
// V[B,S,D] (head-interleaved) -> Vt[B,H,DK,S]
// ---------------------------------------------------------------------------
__global__ __launch_bounds__(256) void transpose_v_kernel(const __bf16* __restrict__ v,
                                                          __bf16* __restrict__ vt) {
    size_t i = (size_t)blockIdx.x * 256 + threadIdx.x;   // over B*S*D
    int d = (int)(i & (cD - 1));
    size_t row = i >> 10;                                 // b*S + s
    int b = (int)(row >> 11);
    int s = (int)(row & (cS - 1));
    int h  = d >> 6;
    int dk = d & (cDK - 1);
    vt[(((size_t)b * cH + h) * cDK + dk) * cS + s] = v[i];
}

// ---------------------------------------------------------------------------
// GEMM: C[M,N] = A[M,K] @ W[N,K]^T + bias.  Block = 8 waves, tile 128M x 64N.
// Weight tile (64N x 64K bf16 = 8KB) staged to LDS by the Tensor Data Mover,
// double-buffered; all 8 waves consume it via ds_load_b128 (8x VMEM reuse).
// ---------------------------------------------------------------------------
template <bool RELU, bool OUT_BF16>
__global__ __launch_bounds__(256) void gemm_kernel(
    const __bf16* __restrict__ A, const __bf16* __restrict__ W,
    const float* __restrict__ bias,
    float* __restrict__ Cf, __bf16* __restrict__ Cb,
    int M, int N, int K) {
    __shared__ __align__(16) __bf16 Bl[2][64][64];

    int lane = threadIdx.x & 31;
    int wv   = threadIdx.x >> 5;
    int half = lane >> 4, l16 = lane & 15;
    int m0 = blockIdx.x * 128 + wv * 16;
    int n0 = blockIdx.y * 64;

    f32x8 acc[4];
#pragma unroll
    for (int t = 0; t < 4; ++t)
#pragma unroll
        for (int r = 0; r < 8; ++r) acc[t][r] = 0.f;

    const __bf16* arow = A + (size_t)(m0 + l16) * K + half * 16;
    const __bf16* wtile = W + (size_t)n0 * K;
    int nstage = K / 64;

    if (wv == 0)
        tdm_load_2d(lds_addr_of(&Bl[0][0][0]), wtile, 64, 64,
                    (unsigned)K, (unsigned)K, (unsigned)N);

    for (int st = 0; st < nstage; ++st) {
        int buf = st & 1;
        if (wv == 0) {
            if (st + 1 < nstage) {
                tdm_load_2d(lds_addr_of(&Bl[buf ^ 1][0][0]), wtile + (st + 1) * 64,
                            64, 64, (unsigned)K, (unsigned)K, (unsigned)N);
                __builtin_amdgcn_s_wait_tensorcnt(1);   // oldest (buf) complete
            } else {
                __builtin_amdgcn_s_wait_tensorcnt(0);
            }
        }
        __syncthreads();                                 // tile visible to all waves
#pragma unroll
        for (int ks = 0; ks < 2; ++ks) {
            bf16x16 af = load_a_frag(arow + st * 64 + ks * 32);
#pragma unroll
            for (int t = 0; t < 4; ++t)
                acc[t] = wmma_bf16(af,
                    load_b_frag(&Bl[buf][t * 16 + l16][ks * 32 + half * 16]),
                    acc[t]);
        }
        __syncthreads();                                 // done reading buf
    }

#pragma unroll
    for (int t = 0; t < 4; ++t) {
        int n = n0 + t * 16 + l16;
        float bv = bias[n];
#pragma unroll
        for (int r = 0; r < 8; ++r) {
            size_t m = (size_t)(m0 + half * 8 + r);
            float v = acc[t][r] + bv;
            if constexpr (RELU) v = fmaxf(v, 0.f);
            if constexpr (OUT_BF16) Cb[m * (size_t)N + n] = (__bf16)v;
            else                    Cf[m * (size_t)N + n] = v;
        }
    }
}

// ---------------------------------------------------------------------------
// Flash attention: grid (S/128, H, B), 8 waves/block, 16 queries per wave.
// K-tile (32x64) and Vt-tile (64x32) staged per 32-key chunk with
// global_load_async_to_lds_b128 (one b128 per thread per tile), then WMMA
// fragments come from LDS. Softmax P relayout bounces through LDS.
// ---------------------------------------------------------------------------
__global__ __launch_bounds__(256) void attn_kernel(
    const __bf16* __restrict__ Q, const __bf16* __restrict__ Kg,
    const __bf16* __restrict__ Vt, const int* __restrict__ mask,
    __bf16* __restrict__ ctx) {
    __shared__ __align__(16) __bf16 Kl[32][64];   // [key in chunk][dk]
    __shared__ __align__(16) __bf16 Vl[64][32];   // [dk][key in chunk]
    __shared__ __align__(16) __bf16 Pl[8][16][32];

    int tid  = threadIdx.x;
    int lane = tid & 31;
    int wv   = tid >> 5;
    int half = lane >> 4, l16 = lane & 15;
    int b = blockIdx.z, h = blockIdx.y;
    int q0 = blockIdx.x * 128 + wv * 16;

    const __bf16* qrow = Q + (size_t)(b * cS + q0 + l16) * cD + h * cDK + half * 16;
    bf16x16 qf0 = load_a_frag(qrow);        // dk 0..31
    bf16x16 qf1 = load_a_frag(qrow + 32);   // dk 32..63

    f32x8 acc[4];
#pragma unroll
    for (int t = 0; t < 4; ++t)
#pragma unroll
        for (int r = 0; r < 8; ++r) acc[t][r] = 0.f;

    float mrun[8], lrun[8];
#pragma unroll
    for (int r = 0; r < 8; ++r) { mrun[r] = -1e30f; lrun[r] = 0.f; }

    const int* mrow = mask + (size_t)b * cS;

    // Per-thread async copy slots (fixed for all chunks)
    int krow = tid >> 3, kch = tid & 7;      // 32 rows x 8 chunks of 8 elems
    int vrow = tid >> 2, vch = tid & 3;      // 64 rows x 4 chunks of 8 elems
    unsigned kdst = lds_addr_of(&Kl[krow][kch * 8]);
    unsigned vdst = lds_addr_of(&Vl[vrow][vch * 8]);
    const __bf16* ksrc0 = Kg + (size_t)(b * cS + krow) * cD + h * cDK + kch * 8;
    const __bf16* vsrc0 = Vt + ((size_t)((b * cH + h) * cDK + vrow)) * cS + vch * 8;

    for (int j0 = 0; j0 < cS; j0 += 32) {
        // ---- stage K/V tiles into LDS (ASYNCcnt path) ----
        async_load_b128(kdst, ksrc0 + (size_t)j0 * cD);
        async_load_b128(vdst, vsrc0 + j0);
        wait_async0();
        __syncthreads();

        // ---- scores: Q(16x64) . K(32x64)^T ----
        f32x8 s0, s1;
#pragma unroll
        for (int r = 0; r < 8; ++r) { s0[r] = 0.f; s1[r] = 0.f; }
        s0 = wmma_bf16(qf0, load_b_frag(&Kl[l16][half * 16]),           s0);
        s0 = wmma_bf16(qf1, load_b_frag(&Kl[l16][32 + half * 16]),      s0);
        s1 = wmma_bf16(qf0, load_b_frag(&Kl[16 + l16][half * 16]),      s1);
        s1 = wmma_bf16(qf1, load_b_frag(&Kl[16 + l16][32 + half * 16]), s1);

        int mv0 = mrow[j0 + l16];
        int mv1 = mrow[j0 + 16 + l16];

#pragma unroll
        for (int r = 0; r < 8; ++r) {
            float a0 = (mv0 == 0) ? -1e9f : s0[r] * 0.125f;   // 1/sqrt(64)
            float a1 = (mv1 == 0) ? -1e9f : s1[r] * 0.125f;
            float cm = fmaxf(a0, a1);
#pragma unroll
            for (int off = 1; off < 16; off <<= 1)
                cm = fmaxf(cm, __shfl_xor(cm, off, 32));
            float mnew = fmaxf(mrun[r], cm);
            float scl  = __expf(mrun[r] - mnew);
            float p0 = __expf(a0 - mnew);
            float p1 = __expf(a1 - mnew);
            float rs = p0 + p1;
#pragma unroll
            for (int off = 1; off < 16; off <<= 1)
                rs += __shfl_xor(rs, off, 32);
            lrun[r] = lrun[r] * scl + rs;
            mrun[r] = mnew;
#pragma unroll
            for (int t = 0; t < 4; ++t) acc[t][r] *= scl;
            Pl[wv][half * 8 + r][l16]      = (__bf16)p0;
            Pl[wv][half * 8 + r][16 + l16] = (__bf16)p1;
        }
        asm volatile("s_wait_dscnt 0x0" ::: "memory");   // intra-wave LDS relayout

        // ---- ctx += P(16x32) . V(32x64) via Vt tile in LDS ----
        bf16x16 pf = load_a_frag(&Pl[wv][l16][half * 16]);
#pragma unroll
        for (int t = 0; t < 4; ++t)
            acc[t] = wmma_bf16(pf, load_b_frag(&Vl[t * 16 + l16][half * 16]), acc[t]);

        __syncthreads();   // all waves done with Kl/Vl before next chunk's DMA
    }

#pragma unroll
    for (int r = 0; r < 8; ++r) {
        float inv = 1.f / lrun[r];
        size_t srow = (size_t)(b * cS + q0 + half * 8 + r);
#pragma unroll
        for (int t = 0; t < 4; ++t)
            ctx[srow * cD + h * cDK + t * 16 + l16] = (__bf16)(acc[t][r] * inv);
    }
}

// ---------------------------------------------------------------------------
// Fused residual + LayerNorm (unbiased std, eps added to std — matches ref).
// ---------------------------------------------------------------------------
__global__ __launch_bounds__(256) void ln_kernel(
    const float* __restrict__ a, const float* __restrict__ res,
    const float* __restrict__ gamma, const float* __restrict__ beta,
    float* __restrict__ outf, __bf16* __restrict__ outb) {
    __shared__ float red[16];
    size_t row = blockIdx.x;
    const float* ap = a + row * cD;
    const float* rp = res + row * cD;

    float v[4];
    float sum = 0.f, sq = 0.f;
#pragma unroll
    for (int i = 0; i < 4; ++i) {
        int c = threadIdx.x + i * 256;
        float t = ap[c] + rp[c];
        v[i] = t;
        sum += t;
        sq = fmaf(t, t, sq);
    }
#pragma unroll
    for (int off = 1; off < 32; off <<= 1) {
        sum += __shfl_xor(sum, off, 32);
        sq  += __shfl_xor(sq,  off, 32);
    }
    int lane = threadIdx.x & 31, wv = threadIdx.x >> 5;
    if (lane == 0) { red[wv] = sum; red[8 + wv] = sq; }
    __syncthreads();
    sum = 0.f; sq = 0.f;
#pragma unroll
    for (int w = 0; w < 8; ++w) { sum += red[w]; sq += red[8 + w]; }

    float mean = sum * (1.f / cD);
    float var  = (sq - (float)cD * mean * mean) * (1.f / (cD - 1));  // ddof=1
    var = var < 0.f ? 0.f : var;
    float inv = 1.f / (sqrtf(var) + cEPS);                            // eps on std
#pragma unroll
    for (int i = 0; i < 4; ++i) {
        int c = threadIdx.x + i * 256;
        float y = gamma[c] * (v[i] - mean) * inv + beta[c];
        outf[row * cD + c] = y;
        if (outb) outb[row * cD + c] = (__bf16)y;
    }
}

// ---------------------------------------------------------------------------
// Host orchestration
// ---------------------------------------------------------------------------
extern "C" void kernel_launch(void* const* d_in, const int* in_sizes, int n_in,
                              void* d_out, int out_size, void* d_ws, size_t ws_size,
                              hipStream_t stream) {
    (void)in_sizes; (void)n_in; (void)out_size; (void)ws_size;

    const float* x    = (const float*)d_in[0];
    const int*   mask = (const int*)  d_in[1];
    const float* Wq   = (const float*)d_in[2];
    const float* bq   = (const float*)d_in[3];
    const float* Wk   = (const float*)d_in[4];
    const float* bk   = (const float*)d_in[5];
    const float* Wv   = (const float*)d_in[6];
    const float* bv   = (const float*)d_in[7];
    const float* Wo   = (const float*)d_in[8];
    const float* bo   = (const float*)d_in[9];
    const float* W1   = (const float*)d_in[10];
    const float* b1   = (const float*)d_in[11];
    const float* W2   = (const float*)d_in[12];
    const float* b2   = (const float*)d_in[13];
    const float* gm1  = (const float*)d_in[14];
    const float* bt1  = (const float*)d_in[15];
    const float* gm2  = (const float*)d_in[16];
    const float* bt2  = (const float*)d_in[17];

    char* wp = (char*)d_ws;
    auto carve = [&](size_t bytes) -> void* {
        void* r = (void*)wp;
        wp += (bytes + 255) & ~(size_t)255;
        return r;
    };

    const size_t nBSD = (size_t)cM * cD;
    __bf16* xb   = (__bf16*)carve(nBSD * 2);
    __bf16* Wqb  = (__bf16*)carve((size_t)cD * cD * 2);
    __bf16* Wkb  = (__bf16*)carve((size_t)cD * cD * 2);
    __bf16* Wvb  = (__bf16*)carve((size_t)cD * cD * 2);
    __bf16* Wob  = (__bf16*)carve((size_t)cD * cD * 2);
    __bf16* W1b  = (__bf16*)carve((size_t)cDFF * cD * 2);
    __bf16* W2b  = (__bf16*)carve((size_t)cD * cDFF * 2);
    __bf16* Qb   = (__bf16*)carve(nBSD * 2);
    __bf16* Kb   = (__bf16*)carve(nBSD * 2);
    __bf16* Vb   = (__bf16*)carve(nBSD * 2);
    __bf16* Vtb  = (__bf16*)carve(nBSD * 2);
    __bf16* ctxb = (__bf16*)carve(nBSD * 2);
    float*  attnf= (float*) carve(nBSD * 4);
    float*  x1f  = (float*) carve(nBSD * 4);
    __bf16* x1b  = (__bf16*)carve(nBSD * 2);
    __bf16* hb   = (__bf16*)carve((size_t)cM * cDFF * 2);
    float*  ffnf = (float*) carve(nBSD * 4);

    cast_kernel<<<(int)(nBSD / 256), 256, 0, stream>>>(x, xb);
    cast_kernel<<<cD * cD / 256, 256, 0, stream>>>(Wq, Wqb);
    cast_kernel<<<cD * cD / 256, 256, 0, stream>>>(Wk, Wkb);
    cast_kernel<<<cD * cD / 256, 256, 0, stream>>>(Wv, Wvb);
    cast_kernel<<<cD * cD / 256, 256, 0, stream>>>(Wo, Wob);
    cast_kernel<<<cDFF * cD / 256, 256, 0, stream>>>(W1, W1b);
    cast_kernel<<<cD * cDFF / 256, 256, 0, stream>>>(W2, W2b);

    dim3 gP(cM / 128, cD / 64);
    gemm_kernel<false, true><<<gP, 256, 0, stream>>>(xb, Wqb, bq, nullptr, Qb, cM, cD, cD);
    gemm_kernel<false, true><<<gP, 256, 0, stream>>>(xb, Wkb, bk, nullptr, Kb, cM, cD, cD);
    gemm_kernel<false, true><<<gP, 256, 0, stream>>>(xb, Wvb, bv, nullptr, Vb, cM, cD, cD);

    transpose_v_kernel<<<(int)(nBSD / 256), 256, 0, stream>>>(Vb, Vtb);

    attn_kernel<<<dim3(cS / 128, cH, cB), 256, 0, stream>>>(Qb, Kb, Vtb, mask, ctxb);

    gemm_kernel<false, false><<<gP, 256, 0, stream>>>(ctxb, Wob, bo, attnf, nullptr, cM, cD, cD);

    ln_kernel<<<cM, 256, 0, stream>>>(x, attnf, gm1, bt1, x1f, x1b);

    dim3 gF1(cM / 128, cDFF / 64);
    gemm_kernel<true, true><<<gF1, 256, 0, stream>>>(x1b, W1b, b1, nullptr, hb, cM, cDFF, cD);
    gemm_kernel<false, false><<<gP, 256, 0, stream>>>(hb, W2b, b2, ffnf, nullptr, cM, cD, cDFF);

    ln_kernel<<<cM, 256, 0, stream>>>(x1f, ffnf, gm2, bt2, (float*)d_out, nullptr);
}
